// GRUCell_10436770529347
// MI455X (gfx1250) — compile-verified, matrix operand order checked by reading, not run on hardware
//
#include <hip/hip_runtime.h>
#include <hip/hip_bf16.h>
#include <math.h>

#define T_DIM 512
#define B_DIM 64
#define I_DIM 512
#define H_DIM 512

typedef __attribute__((ext_vector_type(16))) __bf16 v16bf;
typedef __attribute__((ext_vector_type(8)))  float  v8f;
typedef __attribute__((ext_vector_type(4)))  float  f4;

// ---------------------------------------------------------------------------
// Convert the two big weight matrices (h_w_w [H,I], h_u_w [H,H]) to bf16 once.
// ---------------------------------------------------------------------------
__global__ void cvt_weights_kernel(const float* __restrict__ hww,
                                   const float* __restrict__ huw,
                                   __bf16* __restrict__ hww_b,
                                   __bf16* __restrict__ huw_b, int n) {
  int i = blockIdx.x * blockDim.x + threadIdx.x;
  if (i < n) {
    hww_b[i] = (__bf16)hww[i];
    huw_b[i] = (__bf16)huw[i];
  }
}

// ---------------------------------------------------------------------------
// xz[t,b] = input[t,b,:].zt_w_w + zt_w_b ; xr likewise.  One wave per row.
// ---------------------------------------------------------------------------
__global__ __launch_bounds__(256) void xproj_kernel(
    const float* __restrict__ inp,
    const float* __restrict__ ztw, const float* __restrict__ ztb,
    const float* __restrict__ rtw, const float* __restrict__ rtb,
    float* __restrict__ xz, float* __restrict__ xr) {
  int lane = threadIdx.x & 31;
  int wid  = threadIdx.x >> 5;
  int row  = blockIdx.x * 8 + wid;               // 8 waves per block
  const float* ip = inp + (size_t)row * I_DIM;
  float az = 0.f, ar = 0.f;
  for (int k = lane; k < I_DIM; k += 32) {
    float x = ip[k];
    az += x * ztw[k];
    ar += x * rtw[k];
  }
  for (int off = 16; off > 0; off >>= 1) {
    az += __shfl_xor(az, off, 32);
    ar += __shfl_xor(ar, off, 32);
  }
  if (lane == 0) {
    xz[row] = az + ztb[0];
    xr[row] = ar + rtb[0];
  }
}

// ---------------------------------------------------------------------------
// xn = input[TB,I] @ h_w_w[H,I]^T + h_w_b  -> written straight into d_out.
// Block tile 128x128 (8 waves, each 64x32 = 4x2 accumulators).
// A loaded f32 + converted to bf16 on the fly; B pre-converted bf16.
// ---------------------------------------------------------------------------
__global__ __launch_bounds__(256) void xn_gemm_kernel(
    const float* __restrict__ inp, const __bf16* __restrict__ wb,
    const float* __restrict__ bias, float* __restrict__ out) {
  int lane = threadIdx.x & 31;
  int wid  = threadIdx.x >> 5;
  int wm = wid & 1;           // 0..1 -> 64-row half
  int wn = wid >> 1;          // 0..3 -> 32-col quarter
  int rowBase = blockIdx.x * 128 + wm * 64;
  int colBase = blockIdx.y * 128 + wn * 32;
  int lr = lane & 15;
  int lh = lane >> 4;

  v8f zero = {};
  v8f acc[4][2];
#pragma unroll
  for (int i = 0; i < 4; ++i)
#pragma unroll
    for (int j = 0; j < 2; ++j) acc[i][j] = zero;

  for (int k = 0; k < I_DIM; k += 32) {
    int kb = k + 16 * lh;
    // prefetch next K-tile of weights toward the WGP (global_prefetch_b8)
    if (k + 32 < I_DIM) {
#pragma unroll
      for (int j = 0; j < 2; ++j) {
        int col = colBase + j * 16 + lr;
        __builtin_prefetch(wb + (size_t)col * I_DIM + kb + 32, 0, 1);
      }
    }
    v16bf bfrag[2];
#pragma unroll
    for (int j = 0; j < 2; ++j) {
      int col = colBase + j * 16 + lr;
      bfrag[j] = *(const v16bf*)(wb + (size_t)col * I_DIM + kb);
    }
#pragma unroll
    for (int i = 0; i < 4; ++i) {
      int row = rowBase + i * 16 + lr;
      const f4* ap = (const f4*)(inp + (size_t)row * I_DIM + kb);
      v16bf af;
#pragma unroll
      for (int q = 0; q < 4; ++q) {
        f4 v = ap[q];
        af[q * 4 + 0] = (__bf16)v.x;
        af[q * 4 + 1] = (__bf16)v.y;
        af[q * 4 + 2] = (__bf16)v.z;
        af[q * 4 + 3] = (__bf16)v.w;
      }
#pragma unroll
      for (int j = 0; j < 2; ++j)
        acc[i][j] = __builtin_amdgcn_wmma_f32_16x16x32_bf16(
            false, af, false, bfrag[j], (short)0, acc[i][j], false, false);
    }
  }

#pragma unroll
  for (int i = 0; i < 4; ++i)
#pragma unroll
    for (int j = 0; j < 2; ++j) {
      int col = colBase + j * 16 + lr;
      float bv = bias[col];
#pragma unroll
      for (int v = 0; v < 8; ++v) {
        int row = rowBase + i * 16 + v + 8 * lh;
        out[(size_t)row * H_DIM + col] = acc[i][j][v] + bv;
      }
    }
}

// ---------------------------------------------------------------------------
// One recurrent step. Grid = 16 blocks x 256 threads; block n owns 32 columns.
// Gates zt/rt computed redundantly per block (cheap, L2-hot), broadcast in LDS.
// Step GEMM h[64,512] @ h_u_w[512,512]^T via one 16x16 WMMA tile per wave.
// Reads xn_t from d_out[t], overwrites with h_t in place; also writes hnext.
// ---------------------------------------------------------------------------
__global__ __launch_bounds__(256) void gru_step_kernel(
    const float* __restrict__ hprev, float* __restrict__ hnext,
    const float* __restrict__ xz_t, const float* __restrict__ xr_t,
    const float* __restrict__ ztu, const float* __restrict__ ztub,
    const float* __restrict__ rtu, const float* __restrict__ rtub,
    const __bf16* __restrict__ huw, const float* __restrict__ hub,
    float* __restrict__ out_t) {
  __shared__ float zs[B_DIM];
  __shared__ float rs[B_DIM];

  int tid = threadIdx.x;
  // ---- gates: 4 threads per batch row, 128 elements each (f4 vectorized) ----
  {
    int b = tid >> 2, part = tid & 3;
    const f4* hp = (const f4*)(hprev + (size_t)b * H_DIM + part * 128);
    const f4* zp = (const f4*)(ztu + part * 128);
    const f4* rp = (const f4*)(rtu + part * 128);
    float az = 0.f, ar = 0.f;
#pragma unroll 8
    for (int q = 0; q < 32; ++q) {
      f4 h4 = hp[q], z4 = zp[q], r4 = rp[q];
      az += h4.x * z4.x + h4.y * z4.y + h4.z * z4.z + h4.w * z4.w;
      ar += h4.x * r4.x + h4.y * r4.y + h4.z * r4.z + h4.w * r4.w;
    }
    az += __shfl_xor(az, 1, 32); az += __shfl_xor(az, 2, 32);
    ar += __shfl_xor(ar, 1, 32); ar += __shfl_xor(ar, 2, 32);
    if (part == 0) {
      zs[b] = 1.f / (1.f + __expf(-(xz_t[b] + az + ztub[0])));
      rs[b] = 1.f / (1.f + __expf(-(xr_t[b] + ar + rtub[0])));
    }
  }
  __syncthreads();

  // ---- step GEMM: one 16x16 output tile per wave, K = 512 ----
  int lane = tid & 31, wid = tid >> 5;
  int mi = wid & 3;                 // 4 row-tiles of 16 (B=64)
  int nj = wid >> 2;                // 2 col-tiles of 16
  int lr = lane & 15, lh = lane >> 4;
  int colBase = blockIdx.x * 32 + nj * 16;
  int rowBase = mi * 16;

  v8f acc = {};
  int arow = rowBase + lr;          // A row for this lane
  int bcol = colBase + lr;          // B (weight) row for this lane
  for (int k = 0; k < H_DIM; k += 32) {
    int kb = k + 16 * lh;
    if (k + 32 < H_DIM) {
      __builtin_prefetch(huw + (size_t)bcol * H_DIM + kb + 32, 0, 1);
      __builtin_prefetch(hprev + (size_t)arow * H_DIM + kb + 32, 0, 1);
    }
    v16bf bf = *(const v16bf*)(huw + (size_t)bcol * H_DIM + kb);
    const f4* ap = (const f4*)(hprev + (size_t)arow * H_DIM + kb);
    v16bf af;
#pragma unroll
    for (int q = 0; q < 4; ++q) {
      f4 v = ap[q];
      af[q * 4 + 0] = (__bf16)v.x;
      af[q * 4 + 1] = (__bf16)v.y;
      af[q * 4 + 2] = (__bf16)v.z;
      af[q * 4 + 3] = (__bf16)v.w;
    }
    acc = __builtin_amdgcn_wmma_f32_16x16x32_bf16(
        false, af, false, bf, (short)0, acc, false, false);
  }

  // ---- gate fusion + update ----
  int col = colBase + lr;
  float bv = hub[col];
#pragma unroll
  for (int v = 0; v < 8; ++v) {
    int row = rowBase + v + 8 * lh;
    float g  = acc[v] + bv;
    float z  = zs[row];
    float r  = rs[row];
    size_t idx = (size_t)row * H_DIM + col;
    float xn = out_t[idx];
    float nt = tanhf(xn + g * r);
    float hn = (1.f - z) * nt + z * hprev[idx];
    out_t[idx]  = hn;
    hnext[idx]  = hn;
  }
}

// ---------------------------------------------------------------------------
extern "C" void kernel_launch(void* const* d_in, const int* in_sizes, int n_in,
                              void* d_out, int out_size, void* d_ws, size_t ws_size,
                              hipStream_t stream) {
  const float* inp    = (const float*)d_in[0];   // [T,B,I]
  const float* hidden = (const float*)d_in[1];   // [B,H]
  const float* ztww   = (const float*)d_in[2];   // [1,I]
  const float* ztwb   = (const float*)d_in[3];   // [1]
  const float* ztuw   = (const float*)d_in[4];   // [1,H]
  const float* ztub   = (const float*)d_in[5];   // [1]
  const float* rtww   = (const float*)d_in[6];
  const float* rtwb   = (const float*)d_in[7];
  const float* rtuw   = (const float*)d_in[8];
  const float* rtub   = (const float*)d_in[9];
  const float* hww    = (const float*)d_in[10];  // [H,I]
  const float* hwb    = (const float*)d_in[11];  // [H]
  const float* huw    = (const float*)d_in[12];  // [H,H]
  const float* hub    = (const float*)d_in[13];  // [H]
  float* out = (float*)d_out;                    // [T,B,H] + [B,H]

  // workspace layout (≈1.6 MB)
  char* ws = (char*)d_ws;
  float*  xz    = (float*)(ws);                       // T*B f32      (128KB)
  float*  xr    = (float*)(ws + (131072));            // T*B f32      (128KB)
  __bf16* hww_b = (__bf16*)(ws + (262144));           // H*I bf16     (512KB)
  __bf16* huw_b = (__bf16*)(ws + (786432));           // H*H bf16     (512KB)
  float*  h0    = (float*)(ws + (1310720));           // B*H f32      (128KB)
  float*  h1    = (float*)(ws + (1441792));           // B*H f32      (128KB)

  hipMemcpyAsync(h0, hidden, (size_t)B_DIM * H_DIM * sizeof(float),
                 hipMemcpyDeviceToDevice, stream);

  cvt_weights_kernel<<<(H_DIM * I_DIM) / 256, 256, 0, stream>>>(
      hww, huw, hww_b, huw_b, H_DIM * I_DIM);

  xproj_kernel<<<(T_DIM * B_DIM) / 8, 256, 0, stream>>>(
      inp, ztww, ztwb, rtww, rtwb, xz, xr);

  dim3 ggrid(T_DIM * B_DIM / 128, H_DIM / 128);
  xn_gemm_kernel<<<ggrid, 256, 0, stream>>>(inp, hww_b, hwb, out);

  for (int t = 0; t < T_DIM; ++t) {
    float* hp = (t & 1) ? h1 : h0;
    float* hn = (t & 1) ? h0 : h1;
    gru_step_kernel<<<H_DIM / 32, 256, 0, stream>>>(
        hp, hn, xz + (size_t)t * B_DIM, xr + (size_t)t * B_DIM,
        ztuw, ztub, rtuw, rtub, huw_b, hub,
        out + (size_t)t * B_DIM * H_DIM);
  }
  // T even -> final hidden lives in h0
  hipMemcpyAsync(out + (size_t)T_DIM * B_DIM * H_DIM, h0,
                 (size_t)B_DIM * H_DIM * sizeof(float),
                 hipMemcpyDeviceToDevice, stream);
}
